// CGNNBlock_87196426043567
// MI455X (gfx1250) — matrix operand order repaired and checked by reading, not compiled, not run on hardware
//
#include <hip/hip_runtime.h>
#include <hip/hip_bf16.h>
#include <math.h>

// ---------------------------------------------------------------------------
// CDNA5 / gfx1250 implementation of the CGNN block.
// GEMMs: v_wmma_f32_16x16x32_f16 (fp32 -> f16 staged through LDS, f32 accum).
// LDS layouts are arranged so every A/B fragment is a contiguous 32B read
// (2x ds_load_b128 per fragment), and global staging is float4 (b128) loads.
// Everything else: wave32 row kernels + atomics.
// ---------------------------------------------------------------------------

typedef __attribute__((ext_vector_type(16))) _Float16 v16h;
typedef __attribute__((ext_vector_type(8)))  _Float16 v8h;
typedef __attribute__((ext_vector_type(8)))  float    v8f;

#define DEV __device__ __forceinline__

DEV float gelu_erf(float x) {
  return 0.5f * x * (1.0f + erff(x * 0.70710678118654752f));
}

DEV float wsum(float v) {
#pragma unroll
  for (int off = 16; off > 0; off >>= 1) v += __shfl_xor(v, off, 32);
  return v;
}

// order-preserving float <-> uint encoding for atomicMax
DEV unsigned fenc(float f) {
  unsigned u = __float_as_uint(f);
  return (u & 0x80000000u) ? ~u : (u | 0x80000000u);
}
DEV float fdec(unsigned u) {
  unsigned b = (u & 0x80000000u) ? (u ^ 0x80000000u) : ~u;
  return __uint_as_float(b);
}

DEV v8h cvt8(float4 a, float4 b) {
  v8h r;
  r[0] = (_Float16)a.x; r[1] = (_Float16)a.y;
  r[2] = (_Float16)a.z; r[3] = (_Float16)a.w;
  r[4] = (_Float16)b.x; r[5] = (_Float16)b.y;
  r[6] = (_Float16)b.z; r[7] = (_Float16)b.w;
  return r;
}
DEV v8h zero8() {
  v8h r;
#pragma unroll
  for (int i = 0; i < 8; ++i) r[i] = (_Float16)0.0f;
  return r;
}

// ---------------------------------------------------------------------------
// WMMA GEMM: C[M,N] = act(A[M,K] @ W[K,N] + bias)
// Block: 256 threads (8 waves). Tile: BM=128, BN=64, BK=32.
// Each wave owns a 32x32 sub-tile = 2x2 wmma(16x16x32) accumulators.
//
// LDS layouts:
//  As[row][48]: per-row K order contiguous [0..31], so a lane's
//    A fragment (contiguous K: lanes 0-15 get K0..15; lanes 16-31 get
//    K16..31, same order as Bt) is 16 contiguous halves at half*16.
//  Bt[n][48]: transposed (n-major), so a lane's B fragment (K0..15 or K16..31
//    at fixed n) is 16 contiguous halves at half*16.
// ---------------------------------------------------------------------------
template <int ACT>
__global__ __launch_bounds__(256) void gemm_wmma(
    const float* __restrict__ A, const float* __restrict__ W,
    const float* __restrict__ bias, float* __restrict__ C,
    int M, int N, int K)
{
  __shared__ __align__(32) _Float16 As[128][48];
  __shared__ __align__(32) _Float16 Bt[64][48];

  const int tid  = threadIdx.x;
  const int lane = tid & 31;
  const int wv   = tid >> 5;
  const int bM = blockIdx.x * 128;
  const int bN = blockIdx.y * 64;
  const int wM = (wv & 3) * 32;
  const int wN = (wv >> 2) * 32;
  const int half = lane >> 4;   // lane half selects fragment K-group
  const int l16  = lane & 15;

  v8f acc[2][2] = {};

  const int ar    = tid >> 1;        // A stage: 2 threads/row, 16 floats each
  const int asel  = tid & 1;         // 0 -> K0..15, 1 -> K16..31
  const int p0    = asel ? 16 : 0;   // LDS positions (halves), K order
  const int p1    = asel ? 24 : 8;
  const int br    = tid >> 3;        // B stage: K row
  const int bc    = (tid & 7) * 8;   // B stage: 8-wide N chunk

  for (int k0 = 0; k0 < K; k0 += 32) {
    // --- stage A tile: unconditional float4 loads, branch per row only ---
    {
      const int gm = bM + ar;
      const float* src = A + (long)gm * K + (k0 + asel * 16);
      if (k0 + 32 < K) __builtin_prefetch(src + 32, 0, 1);  // global_prefetch_b8
      v8h lo, hi;
      if (gm < M) {
        float4 f0 = *(const float4*)(src + 0);
        float4 f1 = *(const float4*)(src + 4);
        float4 f2 = *(const float4*)(src + 8);
        float4 f3 = *(const float4*)(src + 12);
        lo = cvt8(f0, f1);
        hi = cvt8(f2, f3);
      } else {
        lo = zero8();
        hi = zero8();
      }
      *(v8h*)&As[ar][p0] = lo;   // 16B ds_store_b128
      *(v8h*)&As[ar][p1] = hi;
    }
    // --- stage B tile transposed: branch per 8-col chunk only ---
    {
      const float* src = W + (long)(k0 + br) * N + (bN + bc);
      if (k0 + 32 < K) __builtin_prefetch(src + 32 * N, 0, 1);
      v8h h;
      if (bN + bc < N) {   // N and bc are multiples of 8: chunk all-in/all-out
        float4 f0 = *(const float4*)(src + 0);
        float4 f1 = *(const float4*)(src + 4);
        h = cvt8(f0, f1);
      } else {
        h = zero8();
      }
#pragma unroll
      for (int t = 0; t < 8; ++t) Bt[bc + t][br] = h[t];
    }
    __syncthreads();

    // --- fragments: one contiguous 32B LDS read each (2x ds_load_b128) ---
    v16h af[2], bf[2];
#pragma unroll
    for (int i = 0; i < 2; ++i)
      af[i] = *(const v16h*)&As[wM + i * 16 + l16][half * 16];
#pragma unroll
    for (int j = 0; j < 2; ++j)
      bf[j] = *(const v16h*)&Bt[wN + j * 16 + l16][half * 16];

#pragma unroll
    for (int i = 0; i < 2; ++i)
#pragma unroll
      for (int j = 0; j < 2; ++j)
        acc[i][j] = __builtin_amdgcn_wmma_f32_16x16x32_f16(
            false, af[i], false, bf[j], (short)0, acc[i][j], false, false);
    __syncthreads();
  }

  // --- epilogue: bias + optional GELU; C layout VGPR r -> M=r / r+8 ---
#pragma unroll
  for (int i = 0; i < 2; ++i) {
#pragma unroll
    for (int j = 0; j < 2; ++j) {
      const int gn = bN + wN + j * 16 + l16;
      if (gn < N) {
        float bv = bias ? bias[gn] : 0.0f;
#pragma unroll
        for (int r = 0; r < 8; ++r) {
          const int gm = bM + wM + i * 16 + half * 8 + r;
          if (gm < M) {
            float v = acc[i][j][r] + bv;
            if (ACT == 1) v = gelu_erf(v);
            C[(long)gm * N + gn] = v;
          }
        }
      }
    }
  }
}

// ---------------------------------------------------------------------------
// LayerNorm over D=128, wave per row. Optional residuals r1,r2; optional
// per-row mean-divide (scatter-mean counts).  In-place safe (x == y).
// ---------------------------------------------------------------------------
__global__ __launch_bounds__(256) void ln128(
    const float* __restrict__ x, const float* __restrict__ r1,
    const float* __restrict__ r2, const float* __restrict__ cnt,
    const float* __restrict__ g, const float* __restrict__ b,
    float* __restrict__ y, int M)
{
  const int row = blockIdx.x * 8 + (threadIdx.x >> 5);
  if (row >= M) return;
  const int lane = threadIdx.x & 31;
  const long base = (long)row * 128;
  float4 v = *(const float4*)(x + base + lane * 4);
  if (r1) { float4 t = *(const float4*)(r1 + base + lane * 4);
            v.x += t.x; v.y += t.y; v.z += t.z; v.w += t.w; }
  if (r2) { float4 t = *(const float4*)(r2 + base + lane * 4);
            v.x += t.x; v.y += t.y; v.z += t.z; v.w += t.w; }
  if (cnt) { float inv = 1.0f / fmaxf(cnt[row], 1.0f);
             v.x *= inv; v.y *= inv; v.z *= inv; v.w *= inv; }
  const float mean = wsum(v.x + v.y + v.z + v.w) * (1.0f / 128.0f);
  const float dx = v.x - mean, dy = v.y - mean, dz = v.z - mean, dw = v.w - mean;
  const float var = wsum(dx * dx + dy * dy + dz * dz + dw * dw) * (1.0f / 128.0f);
  const float rs = rsqrtf(var + 1e-5f);
  float4 gg = *(const float4*)(g + lane * 4);
  float4 bb = *(const float4*)(b + lane * 4);
  float4 o;
  o.x = dx * rs * gg.x + bb.x;
  o.y = dy * rs * gg.y + bb.y;
  o.z = dz * rs * gg.z + bb.z;
  o.w = dw * rs * gg.w + bb.w;
  *(float4*)(y + base + lane * 4) = o;
}

// ---------------------------------------------------------------------------
// SK combine: y = LN(f + sum_F softmax_F(a)*u_f), F = 2 or 3. Wave per row.
// ---------------------------------------------------------------------------
__global__ __launch_bounds__(256) void sk_combine(
    const float* __restrict__ f, const float* __restrict__ u0,
    const float* __restrict__ u1, const float* __restrict__ u2,
    const float* __restrict__ a, const float* __restrict__ g,
    const float* __restrict__ b, float* __restrict__ y, int M, int F)
{
  const int row = blockIdx.x * 8 + (threadIdx.x >> 5);
  if (row >= M) return;
  const int lane = threadIdx.x & 31;
  const long base = (long)row * 128;
  const long abase = (long)row * F * 128;
  float vals[4];
#pragma unroll
  for (int c = 0; c < 4; ++c) {
    const int d = lane * 4 + c;
    const float l0 = a[abase + d];
    const float l1 = a[abase + 128 + d];
    const float l2 = (F == 3) ? a[abase + 256 + d] : -1e30f;
    const float mx = fmaxf(l0, fmaxf(l1, l2));
    const float e0 = expf(l0 - mx), e1 = expf(l1 - mx);
    const float e2 = (F == 3) ? expf(l2 - mx) : 0.0f;
    const float s = e0 + e1 + e2;
    float w = e0 * u0[base + d] + e1 * u1[base + d];
    if (F == 3) w += e2 * u2[base + d];
    vals[c] = f[base + d] + w / s;
  }
  const float mean = wsum(vals[0] + vals[1] + vals[2] + vals[3]) * (1.0f / 128.0f);
  float sq = 0.0f;
#pragma unroll
  for (int c = 0; c < 4; ++c) { const float d = vals[c] - mean; sq += d * d; }
  const float rs = rsqrtf(wsum(sq) * (1.0f / 128.0f) + 1e-5f);
#pragma unroll
  for (int c = 0; c < 4; ++c) {
    const int d = lane * 4 + c;
    y[base + d] = (vals[c] - mean) * rs * g[d] + b[d];
  }
}

// ------------------------- elementwise / gather ----------------------------
#define GS_LOOP(total) \
  for (long i = (long)blockIdx.x * blockDim.x + threadIdx.x; i < (total); \
       i += (long)gridDim.x * blockDim.x)

__global__ void fill_zero(unsigned* __restrict__ p, long n) {
  GS_LOOP(n) p[i] = 0u;
}

// out[e] = [fa[i0[e]], fm[e], fa[i1[e]]]   (rows x 384)
__global__ void gather_concat3(const float* __restrict__ fa,
    const float* __restrict__ fm, const int* __restrict__ i0,
    const int* __restrict__ i1, float* __restrict__ out, int R)
{
  const long total = (long)R * 384;
  GS_LOOP(total) {
    const int e = (int)(i / 384), d = (int)(i % 384);
    float v;
    if (d < 128)      v = fa[(long)i0[e] * 128 + d];
    else if (d < 256) v = fm[(long)e * 128 + (d - 128)];
    else              v = fa[(long)i1[e] * 128 + (d - 256)];
    out[i] = v;
  }
}

// out[e] = [fc[e], fc[pair(e)]]  (E x 256), pair swaps fwd/rev halves
__global__ void sym_concat(const float* __restrict__ fc, float* __restrict__ out, int E)
{
  const int ne = E / 2;
  const long total = (long)E * 256;
  GS_LOOP(total) {
    const int e = (int)(i / 256), d = (int)(i % 256);
    if (d < 128) out[i] = fc[(long)e * 128 + d];
    else {
      const int pe = (e < ne) ? e + ne : e - ne;
      out[i] = fc[(long)pe * 128 + (d - 128)];
    }
  }
}

// out[t] = [fc[soe0], fc[soe1], fc[soe2]]  (T x 384)
__global__ void tri_gather(const float* __restrict__ fc, const int* __restrict__ soe,
                           float* __restrict__ out, int T)
{
  const long total = (long)T * 384;
  GS_LOOP(total) {
    const int t = (int)(i / 384), d = (int)(i % 384);
    const int e = soe[t * 3 + (d >> 7)];
    out[i] = fc[(long)e * 128 + (d & 127)];
  }
}

__global__ void tri_scatter(const float* __restrict__ ft, const int* __restrict__ soe,
                            float* __restrict__ sum, float* __restrict__ cnt, int T)
{
  const long total = (long)T * 128;
  GS_LOOP(total) {
    const int t = (int)(i / 128), d = (int)(i % 128);
    const int e = soe[t * 3 + 2];
    atomicAdd(&sum[(long)e * 128 + d], ft[i]);
    if (d == 0) atomicAdd(&cnt[e], 1.0f);
  }
}

// GLU: out[m,d] = gelu(h[m,dff+d]) * h[m,d]
__global__ void glu_gelu(const float* __restrict__ h, float* __restrict__ out,
                         int M, int dff)
{
  const long total = (long)M * dff;
  GS_LOOP(total) {
    const int m = (int)(i / dff), d = (int)(i % dff);
    const long rb = (long)m * 2 * dff;
    out[i] = gelu_erf(h[rb + dff + d]) * h[rb + d];
  }
}

// ------------------------- attention pieces -------------------------------
// attn[e,h] = (q[tgt[e],h,:] . k[e,h,:]) / sqrt(32). One wave per (edge,head).
__global__ __launch_bounds__(256) void attn_score(
    const float* __restrict__ q, const float* __restrict__ k,
    const int* __restrict__ tgt, float* __restrict__ attn, int E)
{
  const int wv = threadIdx.x >> 5, lane = threadIdx.x & 31;
  const int e = blockIdx.x * 2 + (wv >> 2);
  if (e >= E) return;
  const int h = wv & 3;
  float p = q[(long)tgt[e] * 128 + h * 32 + lane] * k[(long)e * 128 + h * 32 + lane];
  p = wsum(p);
  if (lane == 0) attn[(long)e * 4 + h] = p * 0.17677669529663687f;
}

__global__ void seg_amax(const float* __restrict__ src, const int* __restrict__ idx,
                         unsigned* __restrict__ mx, long n, int width)
{
  GS_LOOP(n) {
    const long r = i / width, c = i % width;
    atomicMax(&mx[(long)idx[r] * width + c], fenc(src[i]));
  }
}

__global__ void seg_expsum(const float* __restrict__ src, const int* __restrict__ idx,
                           const unsigned* __restrict__ mx, float* __restrict__ ex,
                           float* __restrict__ ssum, long n, int width)
{
  GS_LOOP(n) {
    const long r = i / width, c = i % width;
    const long s = (long)idx[r] * width + c;
    const float e = expf(src[i] - fdec(mx[s]));
    ex[i] = e;
    atomicAdd(&ssum[s], e);
  }
}

__global__ void seg_norm(float* __restrict__ ex, const int* __restrict__ idx,
                         const float* __restrict__ ssum, long n, int width)
{
  GS_LOOP(n) {
    const long r = i / width, c = i % width;
    ex[i] = ex[i] / (ssum[(long)idx[r] * width + c] + 1e-7f);
  }
}

// agg[tgt[e],d] += v[e,d] * w[e, d/32]
__global__ void attn_agg(const float* __restrict__ v, const float* __restrict__ w,
                         const int* __restrict__ tgt, float* __restrict__ agg, int E)
{
  const long total = (long)E * 128;
  GS_LOOP(total) {
    const int e = (int)(i / 128), d = (int)(i % 128);
    atomicAdd(&agg[(long)tgt[e] * 128 + d], v[i] * w[(long)e * 4 + (d >> 5)]);
  }
}

// out[idx[l],d] += a[l,d]*b[l,d]   (graph pooling weighted sum)
__global__ void seg_wsum128(const float* __restrict__ a, const float* __restrict__ b,
                            const int* __restrict__ idx, float* __restrict__ out, int R)
{
  const long total = (long)R * 128;
  GS_LOOP(total) {
    const int l = (int)(i / 128), d = (int)(i % 128);
    atomicAdd(&out[(long)idx[l] * 128 + d], a[i] * b[i]);
  }
}

// out[l] = [fn[l], fg[gid[l]]]  (L x 256)
__global__ void concat_ng(const float* __restrict__ fn, const float* __restrict__ fg,
                          const int* __restrict__ gid, float* __restrict__ out, int L)
{
  const long total = (long)L * 256;
  GS_LOOP(total) {
    const int l = (int)(i / 256), d = (int)(i % 256);
    out[i] = (d < 128) ? fn[(long)l * 128 + d] : fg[(long)gid[l] * 128 + (d - 128)];
  }
}

// ---------------------------------------------------------------------------
// Orchestration
// ---------------------------------------------------------------------------
enum {
  P_conn_W, P_conn_b, P_connN_g, P_connN_b,
  P_cu_W, P_cu_b, P_cuN_g, P_cuN_b,
  P_sym_W, P_sym_b, P_symN_g, P_symN_b,
  P_tri_W, P_tri_b, P_triN_g, P_triN_b,
  P_eskF_g, P_eskF_b, P_eskS_W, P_eskS_b,
  P_eskE_W, P_eskE_b, P_eskU_g, P_eskU_b,
  P_eff1_W, P_eff2_W, P_effN_g, P_effN_b,
  P_q_W, P_q_b, P_k_W, P_k_b,
  P_v_W, P_v_b, P_o_W, P_o_b,
  P_attnN_g, P_attnN_b,
  P_gpk_W, P_gpk_b, P_gpv_W, P_gpv_b,
  P_gpN_g, P_gpN_b,
  P_gu_W, P_gu_b, P_guN_g, P_guN_b,
  P_nskF_g, P_nskF_b, P_nskS_W, P_nskS_b,
  P_nskE_W, P_nskE_b, P_nskU_g, P_nskU_b,
  P_nff1_W, P_nff2_W, P_nffN_g, P_nffN_b, NPARAM
};

static const long PSIZES[NPARAM] = {
  49152,128,128,128,  16384,128,128,128,  32768,128,128,128,
  49152,128,128,128,  128,128,4096,32,    12288,384,128,128,
  65536,32768,128,128, 16384,128,16384,128, 49152,128,16384,128,
  128,128,             16384,128,16384,128, 128,128,
  32768,128,128,128,   128,128,4096,32,    8192,256,128,128,
  65536,32768,128,128
};

extern "C" void kernel_launch(void* const* d_in, const int* in_sizes, int n_in,
                              void* d_out, int out_size, void* d_ws, size_t ws_size,
                              hipStream_t stream) {
  const float* f_node = (const float*)d_in[0];
  const float* f_edge = (const float*)d_in[1];
  const int*   ei     = (const int*)d_in[2];
  const int*   gid    = (const int*)d_in[3];
  const int*   soe    = (const int*)d_in[4];

  const int L = in_sizes[0] / 128;
  const int E = in_sizes[1] / 128;
  const int T = in_sizes[4] / 3;
  const int G = 64;
  const int* ei0 = ei;       // edge_index[0] (tgt)
  const int* ei1 = ei + E;   // edge_index[1] (src)

  // params: either one blob (d_in[5]) or per-param pointers (d_in[5..64])
  const float* P[NPARAM];
  if (n_in >= 5 + NPARAM) {
    for (int i = 0; i < NPARAM; ++i) P[i] = (const float*)d_in[5 + i];
  } else {
    const float* base = (const float*)d_in[5];
    long off = 0;
    for (int i = 0; i < NPARAM; ++i) { P[i] = base + off; off += PSIZES[i]; }
  }

  // workspace carve
  float* ws = (float*)d_ws;
  long cur = 0;
  auto alloc = [&](long n) { float* p = ws + cur; cur += n; return p; };
  float* BIG0 = alloc((long)E * 512);
  float* BIG1 = alloc((long)E * 256);
  float* B0 = alloc((long)E * 128);
  float* B1 = alloc((long)E * 128);
  float* B2 = alloc((long)E * 128);
  float* B3 = alloc((long)E * 128);
  float* B4 = alloc((long)E * 128);
  float* B5 = alloc((long)E * 128);
  float* QB = alloc((long)L * 128);
  float* ATT = alloc((long)E * 4);
  unsigned* MX = (unsigned*)alloc((long)L * 4);
  float* SM  = alloc((long)L * 4);
  float* CNT = alloc((long)E);
  unsigned* GMX = (unsigned*)alloc((long)G * 128);
  float* GSM = alloc((long)G * 128);
  float* FG  = alloc((long)G * 128);
  (void)ws_size; (void)out_size;

  float* OUT_N = (float*)d_out;                      // f_node_new [L,128]
  float* OUT_E = (float*)d_out + (long)L * 128;      // f_edge_new [E,128]

  const dim3 blk(256);
  const dim3 ew(4096);
  auto gg = [](int M, int N) { return dim3((M + 127) / 128, (N + 63) / 64); };
  const int lnE = (E + 7) / 8, lnL = (L + 7) / 8, lnG = (G + 7) / 8;

  // ============================ EdgeUpdate =================================
  // fc = LN(concat(fn[ei0], fe, fn[ei1]) @ conn_W + conn_b)
  gather_concat3<<<ew, blk, 0, stream>>>(f_node, f_edge, ei0, ei1, BIG0, E);
  gemm_wmma<0><<<gg(E, 128), blk, 0, stream>>>(BIG0, P[P_conn_W], P[P_conn_b], B0, E, 128, 384);
  ln128<<<lnE, blk, 0, stream>>>(B0, nullptr, nullptr, nullptr, P[P_connN_g], P[P_connN_b], B1, E);
  // u1 = LN(fc + gelu(fc@cu_W + cu_b))
  gemm_wmma<1><<<gg(E, 128), blk, 0, stream>>>(B1, P[P_cu_W], P[P_cu_b], B0, E, 128, 128);
  ln128<<<lnE, blk, 0, stream>>>(B0, B1, nullptr, nullptr, P[P_cuN_g], P[P_cuN_b], B2, E);
  // u2 = LN(gelu(concat(fc, swap(fc)) @ sym_W + sym_b))
  sym_concat<<<ew, blk, 0, stream>>>(B1, BIG0, E);
  gemm_wmma<1><<<gg(E, 128), blk, 0, stream>>>(BIG0, P[P_sym_W], P[P_sym_b], B0, E, 128, 256);
  ln128<<<lnE, blk, 0, stream>>>(B0, nullptr, nullptr, nullptr, P[P_symN_g], P[P_symN_b], B3, E);
  // u3 = LN(scatter_mean(gelu(fc[soe] @ tri_W + tri_b), soe[:,2]))
  tri_gather<<<ew, blk, 0, stream>>>(B1, soe, BIG0, T);
  gemm_wmma<1><<<gg(T, 128), blk, 0, stream>>>(BIG0, P[P_tri_W], P[P_tri_b], B0, T, 128, 384);
  fill_zero<<<ew, blk, 0, stream>>>((unsigned*)B4, (long)E * 128);
  fill_zero<<<ew, blk, 0, stream>>>((unsigned*)CNT, (long)E);
  tri_scatter<<<ew, blk, 0, stream>>>(B0, soe, B4, CNT, T);
  ln128<<<lnE, blk, 0, stream>>>(B4, nullptr, nullptr, CNT, P[P_triN_g], P[P_triN_b], B4, E);
  // esk: a = softmax over F of LN(u1+u2+u3)@S then @E ; fe = LN(f_edge + sum)
  ln128<<<lnE, blk, 0, stream>>>(B2, B3, B4, nullptr, P[P_eskF_g], P[P_eskF_b], B0, E);
  gemm_wmma<1><<<gg(E, 32), blk, 0, stream>>>(B0, P[P_eskS_W], P[P_eskS_b], B5, E, 32, 128);
  gemm_wmma<0><<<gg(E, 384), blk, 0, stream>>>(B5, P[P_eskE_W], P[P_eskE_b], BIG0, E, 384, 32);
  sk_combine<<<lnE, blk, 0, stream>>>(f_edge, B2, B3, B4, BIG0, P[P_eskU_g], P[P_eskU_b], B0, E, 3);
  // eff: GLU feed-forward -> f_edge_new
  gemm_wmma<0><<<gg(E, 512), blk, 0, stream>>>(B0, P[P_eff1_W], nullptr, BIG0, E, 512, 128);
  glu_gelu<<<ew, blk, 0, stream>>>(BIG0, BIG1, E, 256);
  gemm_wmma<0><<<gg(E, 128), blk, 0, stream>>>(BIG1, P[P_eff2_W], nullptr, B2, E, 128, 256);
  ln128<<<lnE, blk, 0, stream>>>(B2, B0, nullptr, nullptr, P[P_effN_g], P[P_effN_b], OUT_E, E);

  // ============================ NodeUpdate =================================
  // fse = concat(fn[src=ei1], f_edge_new, fn[tgt=ei0])
  gather_concat3<<<ew, blk, 0, stream>>>(f_node, OUT_E, ei1, ei0, BIG0, E);
  gemm_wmma<0><<<gg(L, 128), blk, 0, stream>>>(f_node, P[P_q_W], P[P_q_b], QB, L, 128, 128);
  gemm_wmma<0><<<gg(E, 128), blk, 0, stream>>>(OUT_E, P[P_k_W], P[P_k_b], B2, E, 128, 128);
  gemm_wmma<0><<<gg(E, 128), blk, 0, stream>>>(BIG0, P[P_v_W], P[P_v_b], B3, E, 128, 384);
  attn_score<<<dim3((E + 1) / 2), blk, 0, stream>>>(QB, B2, ei0, ATT, E);
  fill_zero<<<ew, blk, 0, stream>>>(MX, (long)L * 4);
  fill_zero<<<ew, blk, 0, stream>>>((unsigned*)SM, (long)L * 4);
  seg_amax<<<ew, blk, 0, stream>>>(ATT, ei0, MX, (long)E * 4, 4);
  seg_expsum<<<ew, blk, 0, stream>>>(ATT, ei0, MX, ATT, SM, (long)E * 4, 4);
  seg_norm<<<ew, blk, 0, stream>>>(ATT, ei0, SM, (long)E * 4, 4);
  fill_zero<<<ew, blk, 0, stream>>>((unsigned*)QB, (long)L * 128);
  attn_agg<<<ew, blk, 0, stream>>>(B3, ATT, ei0, QB, E);
  gemm_wmma<0><<<gg(L, 128), blk, 0, stream>>>(QB, P[P_o_W], P[P_o_b], B4, L, 128, 128);
  ln128<<<lnL, blk, 0, stream>>>(B4, nullptr, nullptr, nullptr, P[P_attnN_g], P[P_attnN_b], B4, L);
  // graph pooling: per-(graph,column) scatter softmax
  gemm_wmma<0><<<gg(L, 128), blk, 0, stream>>>(f_node, P[P_gpk_W], P[P_gpk_b], B2, L, 128, 128);
  gemm_wmma<0><<<gg(L, 128), blk, 0, stream>>>(f_node, P[P_gpv_W], P[P_gpv_b], B3, L, 128, 128);
  fill_zero<<<ew, blk, 0, stream>>>(GMX, (long)G * 128);
  fill_zero<<<ew, blk, 0, stream>>>((unsigned*)GSM, (long)G * 128);
  fill_zero<<<ew, blk, 0, stream>>>((unsigned*)FG, (long)G * 128);
  seg_amax<<<ew, blk, 0, stream>>>(B2, gid, GMX, (long)L * 128, 128);
  seg_expsum<<<ew, blk, 0, stream>>>(B2, gid, GMX, B0, GSM, (long)L * 128, 128);
  seg_norm<<<ew, blk, 0, stream>>>(B0, gid, GSM, (long)L * 128, 128);
  seg_wsum128<<<ew, blk, 0, stream>>>(B0, B3, gid, FG, L);
  ln128<<<lnG, blk, 0, stream>>>(FG, nullptr, nullptr, nullptr, P[P_gpN_g], P[P_gpN_b], FG, G);
  concat_ng<<<ew, blk, 0, stream>>>(f_node, FG, gid, BIG1, L);
  gemm_wmma<1><<<gg(L, 128), blk, 0, stream>>>(BIG1, P[P_gu_W], P[P_gu_b], B2, L, 128, 256);
  ln128<<<lnL, blk, 0, stream>>>(B2, nullptr, nullptr, nullptr, P[P_guN_g], P[P_guN_b], B2, L);
  // nsk: F=2 over [f_in(B4), fng(B2)]
  ln128<<<lnL, blk, 0, stream>>>(B4, B2, nullptr, nullptr, P[P_nskF_g], P[P_nskF_b], B0, L);
  gemm_wmma<1><<<gg(L, 32), blk, 0, stream>>>(B0, P[P_nskS_W], P[P_nskS_b], B5, L, 32, 128);
  gemm_wmma<0><<<gg(L, 256), blk, 0, stream>>>(B5, P[P_nskE_W], P[P_nskE_b], BIG0, L, 256, 32);
  sk_combine<<<lnL, blk, 0, stream>>>(f_node, B4, B2, nullptr, BIG0, P[P_nskU_g], P[P_nskU_b], B3, L, 2);
  // nff: GLU feed-forward -> f_node_new
  gemm_wmma<0><<<gg(L, 512), blk, 0, stream>>>(B3, P[P_nff1_W], nullptr, BIG0, L, 512, 128);
  glu_gelu<<<ew, blk, 0, stream>>>(BIG0, BIG1, L, 256);
  gemm_wmma<0><<<gg(L, 128), blk, 0, stream>>>(BIG1, P[P_nff2_W], nullptr, B0, L, 128, 256);
  ln128<<<lnL, blk, 0, stream>>>(B0, B3, nullptr, nullptr, P[P_nffN_g], P[P_nffN_b], OUT_N, L);
}
